// Predictor_65068754534705
// MI455X (gfx1250) — matrix operand order repaired and checked by reading, not compiled, not run on hardware
//
#include <hip/hip_runtime.h>
#include <math.h>

// ---------------------------------------------------------------------------
// Types for CDNA5 WMMA (gfx1250, wave32): v_wmma_f32_16x16x32_bf16
// ---------------------------------------------------------------------------
typedef __bf16 bf16_t;
typedef __attribute__((ext_vector_type(8)))  __bf16 v8bf;
typedef __attribute__((ext_vector_type(16))) __bf16 v16bf;
typedef __attribute__((ext_vector_type(8)))  float  v8f;

#define CAT16(a, b) __builtin_shufflevector(a, b, 0,1,2,3,4,5,6,7,8,9,10,11,12,13,14,15)

__device__ __forceinline__ v8f wmma_bf16(v16bf a, v16bf b, v8f c) {
  // 8 args: (neg_a, A, neg_b, B, c_mod, C, reuse_a, reuse_b)
  return __builtin_amdgcn_wmma_f32_16x16x32_bf16(false, a, false, b, (short)0, c, false, false);
}

__device__ __forceinline__ float gelu_tanh(float x) {
  return 0.5f * x * (1.0f + tanhf(0.7978845608028654f * (x + 0.044715f * x * x * x)));
}

// ---------------------------------------------------------------------------
// Elementwise: fp32 -> bf16 convert
// ---------------------------------------------------------------------------
__global__ void __launch_bounds__(256) k_cvt_bf16(const float* __restrict__ in,
                                                  bf16_t* __restrict__ out, int n) {
  int i = blockIdx.x * 256 + threadIdx.x;
  if (i < n) out[i] = (bf16_t)in[i];
}

// ---------------------------------------------------------------------------
// Weight transpose + convert: W fp32 [K,N] -> Wt bf16 [N,K] (K-major for WMMA B-frags)
// ---------------------------------------------------------------------------
__global__ void __launch_bounds__(256) k_transpose_w(const float* __restrict__ W,
                                                     bf16_t* __restrict__ Wt,
                                                     int K, int N) {
  __shared__ float tile[32][33];
  int tx = threadIdx.x, ty = threadIdx.y;
  int n0 = blockIdx.x * 32, k0 = blockIdx.y * 32;
  for (int i = ty; i < 32; i += 8) {
    int kk = k0 + i, nn = n0 + tx;
    tile[i][tx] = (kk < K && nn < N) ? W[(size_t)kk * N + nn] : 0.0f;
  }
  __syncthreads();
  for (int i = ty; i < 32; i += 8) {
    int nn = n0 + i, kk = k0 + tx;
    if (nn < N && kk < K) Wt[(size_t)nn * K + kk] = (bf16_t)tile[tx][i];
  }
}

// ---------------------------------------------------------------------------
// LayerNorm: one wave per token (wave32 shuffle reductions), D = 768
// ---------------------------------------------------------------------------
__global__ void __launch_bounds__(32) k_layernorm(const float* __restrict__ x,
                                                  const float* __restrict__ g,
                                                  const float* __restrict__ b,
                                                  float* __restrict__ outF,
                                                  bf16_t* __restrict__ outB, int D) {
  int row = blockIdx.x, lane = threadIdx.x;
  const float* xr = x + (size_t)row * D;
  float s = 0.0f;
  for (int i = lane; i < D; i += 32) s += xr[i];
  for (int off = 16; off; off >>= 1) s += __shfl_xor(s, off, 32);
  float mu = s / (float)D;
  float v = 0.0f;
  for (int i = lane; i < D; i += 32) { float d = xr[i] - mu; v += d * d; }
  for (int off = 16; off; off >>= 1) v += __shfl_xor(v, off, 32);
  float rstd = rsqrtf(v / (float)D + 1e-5f);
  for (int i = lane; i < D; i += 32) {
    float y = (xr[i] - mu) * rstd * g[i] + b[i];
    if (outF) outF[(size_t)row * D + i] = y;
    if (outB) outB[(size_t)row * D + i] = (bf16_t)y;
  }
}

// ---------------------------------------------------------------------------
// WMMA GEMM: out[M,N] = A[M,K](bf16,row-major) x Wt[N,K](bf16,K-major) + bias
//   optional GELU, optional fp32 residual, fp32 and/or bf16 output.
// Block = 256 threads = 8 waves; wave computes 16 rows x 64 cols (4 WMMA tiles,
// A-fragment reused 4x). Grid = (M/128, N/64). K % 32 == 0.
// ---------------------------------------------------------------------------
__global__ void __launch_bounds__(256) k_gemm(const bf16_t* __restrict__ A,
                                              const bf16_t* __restrict__ Wt,
                                              const float* __restrict__ bias,
                                              const float* __restrict__ res,
                                              float* __restrict__ outF,
                                              bf16_t* __restrict__ outB,
                                              int M, int N, int K, int gelu) {
  int wid  = threadIdx.x >> 5;
  int lane = threadIdx.x & 31;
  int lo = lane & 15, hi = lane >> 4;
  int m0 = blockIdx.x * 128 + wid * 16;
  int n0 = blockIdx.y * 64;

  v8f acc[4] = {};
  const bf16_t* pa = A  + (size_t)(m0 + lo) * K + hi * 16;
  const bf16_t* pb = Wt + (size_t)(n0 + lo) * K + hi * 16;

  for (int k = 0; k < K; k += 32) {
    // A fragment: lanes 0-15 row lo, K = k..k+15; lanes 16-31: +16
    v8bf a0 = *(const v8bf*)(pa + k);
    v8bf a1 = *(const v8bf*)(pa + k + 8);
    v16bf af = CAT16(a0, a1);
#pragma unroll
    for (int t = 0; t < 4; ++t) {
      // B fragment: lane n = n0+t*16+lo, 32B contiguous along K
      v16bf bf = *(const v16bf*)(pb + (size_t)t * 16 * K + k);
      acc[t] = wmma_bf16(af, bf, acc[t]);
    }
  }

#pragma unroll
  for (int t = 0; t < 4; ++t) {
    int col = n0 + t * 16 + lo;
    float bv = bias[col];
#pragma unroll
    for (int j = 0; j < 8; ++j) {
      int row = m0 + j + hi * 8;            // C layout: VGPR j -> M = j + 8*hi
      float v = acc[t][j] + bv;
      if (gelu) v = gelu_tanh(v);
      size_t o = (size_t)row * N + col;
      if (res)  v += res[o];
      if (outF) outF[o] = v;
      if (outB) outB[o] = (bf16_t)v;
    }
  }
}

// ---------------------------------------------------------------------------
// Split/convert attention operands out of fp32 projection outputs:
//   dst (bf16) = [B,H,TPad,64]       (tr==0)  rows t>=Tv zero-padded
//   dst (bf16) = [B,H,64,TPad]       (tr==1)  transposed (for V^T), padded
// ---------------------------------------------------------------------------
__global__ void __launch_bounds__(256) k_split(const float* __restrict__ src,
                                               bf16_t* __restrict__ dst,
                                               int srcStride, int colOff, int Tv,
                                               int TPad, int Hn, float scale,
                                               int tr, int total) {
  int idx = blockIdx.x * 256 + threadIdx.x;
  if (idx >= total) return;
  int per = TPad * 64;
  int bh = idx / per;
  int rem = idx - bh * per;
  int t, d;
  if (tr) { t = rem % TPad; d = rem / TPad; }
  else    { d = rem & 63;   t = rem >> 6;  }
  int b = bh / Hn, h = bh - b * Hn;
  float v = 0.0f;
  if (t < Tv) v = src[((size_t)b * Tv + t) * srcStride + colOff + h * 64 + d] * scale;
  dst[idx] = (bf16_t)v;
}

// ---------------------------------------------------------------------------
// Gather positional queries: out[b,t,:] = table[pos[b,t],:]
// ---------------------------------------------------------------------------
__global__ void __launch_bounds__(256) k_gather(const float* __restrict__ table,
                                                const int* __restrict__ pos,
                                                float* __restrict__ out,
                                                int D, int total) {
  int idx = blockIdx.x * 256 + threadIdx.x;
  if (idx >= total) return;
  int d = idx % D, bt = idx / D;
  out[idx] = table[(size_t)pos[bt] * D + d];
}

// ---------------------------------------------------------------------------
// Fused WMMA attention: one wave per (b,h, 16-query tile).
//   Q [B,H,TqPad,64] bf16 (pre-scaled by DH^-0.5), K [B,H,SkPad,64] bf16,
//   Vt [B,H,64,SkPad] bf16.  S = Q K^T via WMMA -> LDS -> softmax (shuffle)
//   -> P bf16 in LDS -> O = P V via WMMA -> Obf [B,Tq,H*64] bf16.
// ---------------------------------------------------------------------------
#define SKMAX 224
__global__ void __launch_bounds__(32) k_attn(const bf16_t* __restrict__ Qbf,
                                             const bf16_t* __restrict__ Kbf,
                                             const bf16_t* __restrict__ Vt,
                                             bf16_t* __restrict__ Obf,
                                             int Tq, int TqPad, int Sk, int SkPad,
                                             int Hn) {
  __shared__ float  sS[16][SKMAX];
  __shared__ __bf16 sP[16][SKMAX];
  int bh = blockIdx.y;
  int b = bh / Hn, h = bh - b * Hn;
  int qt = blockIdx.x;
  int lane = threadIdx.x;
  int lo = lane & 15, hi = lane >> 4;

  const bf16_t* Qb = Qbf + ((size_t)bh * TqPad + qt * 16) * 64;
  const bf16_t* Kb = Kbf + (size_t)bh * SkPad * 64;
  const bf16_t* Vb = Vt  + (size_t)bh * 64 * SkPad;

  // Q fragments for K-steps 0 and 32 (DH = 64)
  v16bf qf[2];
#pragma unroll
  for (int ki = 0; ki < 2; ++ki) {
    const bf16_t* p = Qb + (size_t)lo * 64 + ki * 32 + hi * 16;
    v8bf a0 = *(const v8bf*)p;
    v8bf a1 = *(const v8bf*)(p + 8);
    qf[ki] = CAT16(a0, a1);
  }

  // Scores: S = Q K^T, tile over keys, dump C tiles to LDS
  int ntiles = SkPad >> 4;
  for (int nt = 0; nt < ntiles; ++nt) {
    v8f acc = {};
#pragma unroll
    for (int ki = 0; ki < 2; ++ki) {
      // B-frag of K^T: lane n holds K-row (nt*16+lo), contiguous along DH
      v16bf bf = *(const v16bf*)(Kb + (size_t)(nt * 16 + lo) * 64 + ki * 32 + hi * 16);
      acc = wmma_bf16(qf[ki], bf, acc);
    }
#pragma unroll
    for (int j = 0; j < 8; ++j) sS[j + hi * 8][nt * 16 + lo] = acc[j];
  }
  __syncthreads();

  // Softmax per row (wave32 cooperates across columns, pad cols masked)
  for (int r = 0; r < 16; ++r) {
    float vals[SKMAX / 32];
    float m = -1e30f;
    int i = 0;
    for (int c = lane; c < SkPad; c += 32, ++i) {
      float v = (c < Sk) ? sS[r][c] : -1e30f;
      vals[i] = v;
      m = fmaxf(m, v);
    }
    for (int off = 16; off; off >>= 1) m = fmaxf(m, __shfl_xor(m, off, 32));
    float s = 0.0f;
    i = 0;
    for (int c = lane; c < SkPad; c += 32, ++i) {
      float e = (c < Sk) ? __expf(vals[i] - m) : 0.0f;
      vals[i] = e;
      s += e;
    }
    for (int off = 16; off; off >>= 1) s += __shfl_xor(s, off, 32);
    float inv = 1.0f / s;
    i = 0;
    for (int c = lane; c < SkPad; c += 32, ++i) sP[r][c] = (bf16_t)(vals[i] * inv);
  }
  __syncthreads();

  // O = P V : K-dim = SkPad (multiple of 32), 4 tiles over DH=64
#pragma unroll
  for (int dt = 0; dt < 4; ++dt) {
    v8f acc = {};
    for (int ks = 0; ks < SkPad; ks += 32) {
      const bf16_t* p = &sP[lo][ks + hi * 16];
      v8bf a0 = *(const v8bf*)p;
      v8bf a1 = *(const v8bf*)(p + 8);
      v16bf af = CAT16(a0, a1);
      v16bf bf = *(const v16bf*)(Vb + (size_t)(dt * 16 + lo) * SkPad + ks + hi * 16);
      acc = wmma_bf16(af, bf, acc);
    }
#pragma unroll
    for (int j = 0; j < 8; ++j) {
      int row = qt * 16 + j + hi * 8;
      if (row < Tq)
        Obf[((size_t)b * Tq + row) * (64 * Hn) + h * 64 + dt * 16 + lo] = (bf16_t)acc[j];
    }
  }
}

// ---------------------------------------------------------------------------
// Host orchestration
// ---------------------------------------------------------------------------
extern "C" void kernel_launch(void* const* d_in, const int* in_sizes, int n_in,
                              void* d_out, int out_size, void* d_ws, size_t ws_size,
                              hipStream_t stream) {
  (void)in_sizes; (void)n_in; (void)out_size; (void)ws_size;
  constexpr int B = 64, S = 196, D = 768, T = 98, H = 12, L = 3, FF = 4 * 768;
  constexpr int MS = B * S;   // 12544 (multiple of 128)
  constexpr int MT = B * T;   // 6272  (multiple of 128)
  constexpr int SPadE = 224, QPadE = 208;   // encoder self-attn pads
  constexpr int SPadD = 128, QPadD = 112;   // decoder pads

  // ---- inputs (setup_inputs dict order) ----
  const float* ctx    = (const float*)d_in[0];
  const int*   tpos   = (const int*)d_in[1];
  const float* spe    = (const float*)d_in[2];
  const float* norm_g = (const float*)d_in[3];
  const float* norm_b = (const float*)d_in[4];
  const float* e_ln1g = (const float*)d_in[5];
  const float* e_ln1b = (const float*)d_in[6];
  const float* e_wqkv = (const float*)d_in[7];
  const float* e_bqkv = (const float*)d_in[8];
  const float* e_wo   = (const float*)d_in[9];
  const float* e_bo   = (const float*)d_in[10];
  const float* e_ln2g = (const float*)d_in[11];
  const float* e_ln2b = (const float*)d_in[12];
  const float* e_w1   = (const float*)d_in[13];
  const float* e_b1   = (const float*)d_in[14];
  const float* e_w2   = (const float*)d_in[15];
  const float* e_b2   = (const float*)d_in[16];
  const float* d_ln1g = (const float*)d_in[17];
  const float* d_ln1b = (const float*)d_in[18];
  const float* d_wqkv = (const float*)d_in[19];
  const float* d_bqkv = (const float*)d_in[20];
  const float* d_wo   = (const float*)d_in[21];
  const float* d_bo   = (const float*)d_in[22];
  const float* d_ln2g = (const float*)d_in[23];
  const float* d_ln2b = (const float*)d_in[24];
  const float* d_w1   = (const float*)d_in[25];
  const float* d_b1   = (const float*)d_in[26];
  const float* d_w2   = (const float*)d_in[27];
  const float* d_b2   = (const float*)d_in[28];
  const float* ca_wq  = (const float*)d_in[29];
  const float* ca_bq  = (const float*)d_in[30];
  const float* ca_wk  = (const float*)d_in[31];
  const float* ca_bk  = (const float*)d_in[32];
  const float* ca_wv  = (const float*)d_in[33];
  const float* ca_bv  = (const float*)d_in[34];
  const float* ca_wo  = (const float*)d_in[35];
  const float* ca_bo  = (const float*)d_in[36];

  // ---- workspace bump allocator ----
  uint8_t* base = (uint8_t*)d_ws;
  size_t off = 0;
  auto alloc = [&](size_t bytes) -> void* {
    void* p = base + off;
    off = (off + bytes + 255) & ~(size_t)255;
    return p;
  };
  bf16_t*  wT   = (bf16_t*)alloc((size_t)FF * D * 2);        // rotating weight^T slot
  float*   xA   = (float*)alloc((size_t)MS * D * 4);
  float*   xB   = (float*)alloc((size_t)MS * D * 4);
  bf16_t*  hbf  = (bf16_t*)alloc((size_t)MS * D * 2);
  uint8_t* big  = (uint8_t*)alloc((size_t)MS * 3 * D * 4);   // qkv fp32 / gelu bf16 / kk,vv,qq
  bf16_t*  Qbf  = (bf16_t*)alloc((size_t)B * H * QPadE * 64 * 2);
  bf16_t*  Kbf  = (bf16_t*)alloc((size_t)B * H * SPadE * 64 * 2);
  bf16_t*  Vtb  = (bf16_t*)alloc((size_t)B * H * 64 * SPadE * 2);
  bf16_t*  Obf  = (bf16_t*)alloc((size_t)MS * D * 2);
  bf16_t*  xebf = (bf16_t*)alloc((size_t)MS * D * 2);
  float*   qxA  = (float*)alloc((size_t)MT * D * 4);
  float*   qxB  = (float*)alloc((size_t)MT * D * 4);
  bf16_t*  qbf  = (bf16_t*)alloc((size_t)MT * D * 2);

  auto transW = [&](const float* W, int K, int N) {
    dim3 g(N / 32, K / 32);
    k_transpose_w<<<g, dim3(32, 8), 0, stream>>>(W, wT, K, N);
  };
  auto gemm = [&](const bf16_t* A, const float* bias, const float* res,
                  float* oF, bf16_t* oB, int M, int N, int K, int gelu) {
    dim3 g(M / 128, N / 64);
    k_gemm<<<g, 256, 0, stream>>>(A, wT, bias, res, oF, oB, M, N, K, gelu);
  };
  auto split = [&](const float* src, bf16_t* dst, int stride, int colOff, int Tv,
                   int TPad, float scale, int tr) {
    int total = B * H * TPad * 64;
    k_split<<<(total + 255) / 256, 256, 0, stream>>>(src, dst, stride, colOff, Tv,
                                                     TPad, H, scale, tr, total);
  };
  auto attn = [&](int Tq, int TqPad, int Sk, int SkPad) {
    dim3 g(TqPad / 16, B * H);
    k_attn<<<g, 32, 0, stream>>>(Qbf, Kbf, Vtb, Obf, Tq, TqPad, Sk, SkPad, H);
  };
  auto ln = [&](const float* x, const float* g, const float* bb, float* oF,
                bf16_t* oB, int M) {
    k_layernorm<<<M, 32, 0, stream>>>(x, g, bb, oF, oB, D);
  };
  auto cvt = [&](const float* in, bf16_t* out, int n) {
    k_cvt_bf16<<<(n + 255) / 256, 256, 0, stream>>>(in, out, n);
  };

  const float SCALE = 0.125f;  // DH^-0.5, folded into Q

  // ======================= Encoder =======================
  hipMemcpyAsync(xA, ctx, (size_t)MS * D * 4, hipMemcpyDeviceToDevice, stream);
  float* x = xA;
  float* y = xB;
  for (int l = 0; l < L; ++l) {
    ln(x, e_ln1g + (size_t)l * D, e_ln1b + (size_t)l * D, nullptr, hbf, MS);
    transW(e_wqkv + (size_t)l * D * 3 * D, D, 3 * D);
    float* qkv = (float*)big;
    gemm(hbf, e_bqkv + (size_t)l * 3 * D, nullptr, qkv, nullptr, MS, 3 * D, D, 0);
    split(qkv, Qbf, 3 * D, 0,     S, QPadE, SCALE, 0);
    split(qkv, Kbf, 3 * D, D,     S, SPadE, 1.0f,  0);
    split(qkv, Vtb, 3 * D, 2 * D, S, SPadE, 1.0f,  1);
    attn(S, QPadE, S, SPadE);
    transW(e_wo + (size_t)l * D * D, D, D);
    gemm(Obf, e_bo + (size_t)l * D, x, y, nullptr, MS, D, D, 0);
    { float* t = x; x = y; y = t; }
    ln(x, e_ln2g + (size_t)l * D, e_ln2b + (size_t)l * D, nullptr, hbf, MS);
    transW(e_w1 + (size_t)l * D * FF, D, FF);
    bf16_t* gb = (bf16_t*)big;
    gemm(hbf, e_b1 + (size_t)l * FF, nullptr, nullptr, gb, MS, FF, D, 1);
    transW(e_w2 + (size_t)l * FF * D, FF, D);
    gemm(gb, e_b2 + (size_t)l * D, x, y, nullptr, MS, D, FF, 0);
    { float* t = x; x = y; y = t; }
  }

  // ==================== Cross-attn + Decoder ====================
  cvt(x, xebf, MS * D);  // encoder output as bf16 A-operand for K/V projections
  k_gather<<<(MT * D + 255) / 256, 256, 0, stream>>>(spe, tpos, qxA, D, MT * D);
  float* q  = qxA;
  float* qa = qxB;
  float* kk = (float*)(big);
  float* vv = (float*)(big + (size_t)MS * D * 4);
  float* qq = (float*)(big + (size_t)2 * MS * D * 4);

  for (int l = 0; l < L; ++l) {
    // --- cross attention (no residual; output replaces q) ---
    cvt(q, qbf, MT * D);
    transW(ca_wq + (size_t)l * D * D, D, D);
    gemm(qbf, ca_bq + (size_t)l * D, nullptr, qq, nullptr, MT, D, D, 0);
    transW(ca_wk + (size_t)l * D * D, D, D);
    gemm(xebf, ca_bk + (size_t)l * D, nullptr, kk, nullptr, MS, D, D, 0);
    transW(ca_wv + (size_t)l * D * D, D, D);
    gemm(xebf, ca_bv + (size_t)l * D, nullptr, vv, nullptr, MS, D, D, 0);
    split(qq, Qbf, D, 0, T, QPadD, SCALE, 0);
    split(kk, Kbf, D, 0, S, SPadE, 1.0f,  0);
    split(vv, Vtb, D, 0, S, SPadE, 1.0f,  1);
    attn(T, QPadD, S, SPadE);
    transW(ca_wo + (size_t)l * D * D, D, D);
    gemm(Obf, ca_bo + (size_t)l * D, nullptr, q, nullptr, MT, D, D, 0);

    // --- decoder self-attn block ---
    ln(q, d_ln1g + (size_t)l * D, d_ln1b + (size_t)l * D, nullptr, hbf, MT);
    transW(d_wqkv + (size_t)l * D * 3 * D, D, 3 * D);
    float* dqkv = (float*)big;
    gemm(hbf, d_bqkv + (size_t)l * 3 * D, nullptr, dqkv, nullptr, MT, 3 * D, D, 0);
    split(dqkv, Qbf, 3 * D, 0,     T, QPadD, SCALE, 0);
    split(dqkv, Kbf, 3 * D, D,     T, SPadD, 1.0f,  0);
    split(dqkv, Vtb, 3 * D, 2 * D, T, SPadD, 1.0f,  1);
    attn(T, QPadD, T, SPadD);
    transW(d_wo + (size_t)l * D * D, D, D);
    gemm(Obf, d_bo + (size_t)l * D, q, qa, nullptr, MT, D, D, 0);
    { float* t = q; q = qa; qa = t; }
    ln(q, d_ln2g + (size_t)l * D, d_ln2b + (size_t)l * D, nullptr, hbf, MT);
    transW(d_w1 + (size_t)l * D * FF, D, FF);
    bf16_t* gb = (bf16_t*)big;
    gemm(hbf, d_b1 + (size_t)l * FF, nullptr, nullptr, gb, MT, FF, D, 1);
    transW(d_w2 + (size_t)l * FF * D, FF, D);
    gemm(gb, d_b2 + (size_t)l * D, q, qa, nullptr, MT, D, FF, 0);
    { float* t = q; q = qa; qa = t; }
  }

  // ======================= Final LayerNorm -> d_out =======================
  ln(q, norm_g, norm_b, (float*)d_out, nullptr, MT);
}
